// GATES_40553081209266
// MI455X (gfx1250) — matrix operand order repaired and checked by reading, not compiled, not run on hardware
//
#include <hip/hip_runtime.h>

// ---------------------------------------------------------------------------
// GAT pipeline for MI455X (gfx1250, wave32, WMMA)
//   x -> bf16 once       -> pure-load GEMM A operand (no per-tile cvt VALU)
//   h = x @ W            -> v_wmma_f32_16x16x32_bf16, fully unrolled K loop
//   scores               -> wave-per-node dot products
//   segment softmax      -> atomicMax (ordered uint) + atomicAdd passes
//   scatter aggregate    -> wave-per-edge, coalesced gather + f32 atomics
// ---------------------------------------------------------------------------

typedef __attribute__((ext_vector_type(16))) __bf16 v16bf;
typedef __attribute__((ext_vector_type(8)))  float  v8f;

#define NNODES   100000
#define K1       512
#define H1       256
#define H2       64
#define NEDGES   800000
#define NEG_SLOPE 0.2f

__device__ __forceinline__ unsigned short f2bf(float f) {
    unsigned u = __float_as_uint(f);
    unsigned r = u + 0x7FFFu + ((u >> 16) & 1u);   // round-to-nearest-even
    return (unsigned short)(r >> 16);
}

// order-preserving float <-> uint mapping for atomicMax on floats
__device__ __forceinline__ unsigned ordf(float f) {
    unsigned u = __float_as_uint(f);
    return (u & 0x80000000u) ? ~u : (u | 0x80000000u);
}
__device__ __forceinline__ float deordf(unsigned e) {
    unsigned u = (e & 0x80000000u) ? (e ^ 0x80000000u) : ~e;
    return __uint_as_float(u);
}

__device__ __forceinline__ float lrelu(float x) {
    return x >= 0.0f ? x : NEG_SLOPE * x;
}

// ---------------------------------------------------------------------------
// f32 -> bf16 row-major conversion (vectorized 4-wide)
// ---------------------------------------------------------------------------
__global__ void cvt_bf16_kernel(const float* __restrict__ in,
                                unsigned short* __restrict__ out, int n4) {
    int i = blockIdx.x * blockDim.x + threadIdx.x;
    if (i >= n4) return;
    float4 f = reinterpret_cast<const float4*>(in)[i];
    ushort4 o;
    o.x = f2bf(f.x); o.y = f2bf(f.y); o.z = f2bf(f.z); o.w = f2bf(f.w);
    reinterpret_cast<ushort4*>(out)[i] = o;
}

// ---------------------------------------------------------------------------
// Pack W [K x Nout] f32 row-major into WMMA B-fragment order (bf16).
// Fragment (kt, nt): lane l holds column n0+(l&15), rows k0+kbase..+15,
// kbase = (l<16)?0:16; element i -> k = k0+kbase+i. One v16bf per lane.
// ---------------------------------------------------------------------------
__global__ void pack_w_kernel(const float* __restrict__ W,
                              unsigned short* __restrict__ Wp,
                              int K, int Nout) {
    int tid = blockIdx.x * blockDim.x + threadIdx.x;
    int Ktiles = K >> 5, Ntiles = Nout >> 4;
    int total = Ktiles * Ntiles * 32;
    if (tid >= total) return;
    int lane = tid & 31;
    int t = tid >> 5;
    int nt = t % Ntiles;
    int kt = t / Ntiles;
    int kbase = (lane < 16) ? 0 : 16;
    int n = nt * 16 + (lane & 15);
    int k0 = kt * 32 + kbase;
#pragma unroll
    for (int i = 0; i < 16; ++i)
        Wp[(size_t)tid * 16 + i] = f2bf(W[(size_t)(k0 + i) * Nout + n]);
}

// ---------------------------------------------------------------------------
// GEMM: C[M x NOUT] = A[M x K] (bf16 row-major) @ Wp (packed bf16 fragments)
// One wave per 16x16 C tile; block = 4 waves covering 4 consecutive N tiles.
// A fragment per lane = two contiguous 16B chunks (k0+kbase .. +7, +8 .. +15).
// K, NOUT compile-time -> fully unrolled WMMA chain.
// ---------------------------------------------------------------------------
template <int K, int NOUT>
__global__ void gemm_bf16_kernel(const unsigned short* __restrict__ A,
                                 const unsigned short* __restrict__ Wp,
                                 float* __restrict__ C) {
    constexpr int Ntiles = NOUT >> 4;
    constexpr int Ktiles = K >> 5;
    const int lane = threadIdx.x & 31;
    const int wave = threadIdx.x >> 5;
    const int row0 = blockIdx.x * 16;
    const int nt = blockIdx.y * 4 + wave;
    const int n0 = nt * 16;

    const int m      = lane & 15;
    const int kbaseA = (lane < 16) ? 0 : 16;   // A-matrix 16x32 bf16 layout

    v8f acc;
#pragma unroll
    for (int r = 0; r < 8; ++r) acc[r] = 0.0f;

    const unsigned short* arow = A + (size_t)(row0 + m) * K + kbaseA;
    const unsigned short* bptr = Wp + ((size_t)nt * 32 + lane) * 16;

#pragma unroll
    for (int kt = 0; kt < Ktiles; ++kt) {
        union { v16bf v; uint4 q[2]; } af;
        af.q[0] = *reinterpret_cast<const uint4*>(arow + kt * 32);       // i 0..7
        af.q[1] = *reinterpret_cast<const uint4*>(arow + kt * 32 + 8);  // i 8..15
        const v16bf b = *reinterpret_cast<const v16bf*>(
            bptr + (size_t)kt * Ntiles * 32 * 16);
        acc = __builtin_amdgcn_wmma_f32_16x16x32_bf16(
            false, af.v, false, b, (short)0, acc, false, false);
    }

    // D layout: lanes 0-15 -> M = r, lanes 16-31 -> M = r+8 ; N = lane&15
    const int mrow = (lane < 16) ? 0 : 8;
    const int n = n0 + (lane & 15);
#pragma unroll
    for (int r = 0; r < 8; ++r)
        C[(size_t)(row0 + r + mrow) * NOUT + n] = acc[r];
}

// ---------------------------------------------------------------------------
// Per-node attention scores: s_src[i] = <h[i], a_src>, s_dst[i] = <h[i], a_dst>
// One wave per node.
// ---------------------------------------------------------------------------
__global__ void scores_kernel(const float* __restrict__ H,
                              const float* __restrict__ a_src,
                              const float* __restrict__ a_dst,
                              float* __restrict__ ssrc,
                              float* __restrict__ sdst,
                              int n_nodes, int Hdim) {
    int node = blockIdx.x * 8 + (threadIdx.x >> 5);
    int lane = threadIdx.x & 31;
    if (node >= n_nodes) return;
    float s0 = 0.0f, s1 = 0.0f;
    const float* h = H + (size_t)node * Hdim;
    for (int d = lane; d < Hdim; d += 32) {
        float hv = h[d];
        s0 += hv * a_src[d];
        s1 += hv * a_dst[d];
    }
#pragma unroll
    for (int off = 16; off > 0; off >>= 1) {
        s0 += __shfl_xor(s0, off);
        s1 += __shfl_xor(s1, off);
    }
    if (lane == 0) { ssrc[node] = s0; sdst[node] = s1; }
}

__global__ void init_max_kernel(unsigned* __restrict__ mord, int n) {
    int i = blockIdx.x * blockDim.x + threadIdx.x;
    if (i < n) mord[i] = 0x007FFFFFu;   // ordf(-inf)
}

__device__ __forceinline__ void edge_nodes(const long long* __restrict__ src,
                                           const long long* __restrict__ dst,
                                           int E, int id, int& s, int& d) {
    if (id < E) { s = (int)src[id]; d = (int)dst[id]; }
    else        { s = d = id - E; }          // self loops
}

__global__ void edge_max_kernel(const long long* __restrict__ src,
                                const long long* __restrict__ dst,
                                int E, int n_nodes,
                                const float* __restrict__ ssrc,
                                const float* __restrict__ sdst,
                                unsigned* __restrict__ mord) {
    int id = blockIdx.x * blockDim.x + threadIdx.x;
    if (id >= E + n_nodes) return;
    int s, d; edge_nodes(src, dst, E, id, s, d);
    float e = lrelu(ssrc[s] + sdst[d]);
    atomicMax(&mord[d], ordf(e));
}

__global__ void edge_sum_kernel(const long long* __restrict__ src,
                                const long long* __restrict__ dst,
                                int E, int n_nodes,
                                const float* __restrict__ ssrc,
                                const float* __restrict__ sdst,
                                const unsigned* __restrict__ mord,
                                float* __restrict__ den) {
    int id = blockIdx.x * blockDim.x + threadIdx.x;
    if (id >= E + n_nodes) return;
    int s, d; edge_nodes(src, dst, E, id, s, d);
    float e = lrelu(ssrc[s] + sdst[d]);
    atomicAdd(&den[d], __expf(e - deordf(mord[d])));
}

// one wave per edge: gather h[src] coalesced, scatter coef*h into agg[dst]
__global__ void edge_scatter_kernel(const long long* __restrict__ src,
                                    const long long* __restrict__ dst,
                                    int E, int n_nodes,
                                    const float* __restrict__ ssrc,
                                    const float* __restrict__ sdst,
                                    const unsigned* __restrict__ mord,
                                    const float* __restrict__ den,
                                    const float* __restrict__ H,
                                    float* __restrict__ agg,
                                    int Hdim) {
    int edge = blockIdx.x * 8 + (threadIdx.x >> 5);
    int lane = threadIdx.x & 31;
    if (edge >= E + n_nodes) return;
    int s, d; edge_nodes(src, dst, E, edge, s, d);
    float coef = 0.0f;
    if (lane == 0) {
        float e = lrelu(ssrc[s] + sdst[d]);
        coef = __expf(e - deordf(mord[d])) / den[d];
    }
    coef = __shfl(coef, 0);
    const float* hs = H + (size_t)s * Hdim;
    float* ad = agg + (size_t)d * Hdim;
    for (int dd = lane; dd < Hdim; dd += 32)
        atomicAdd(&ad[dd], coef * hs[dd]);
}

__global__ void bias_elu_kernel(float* __restrict__ agg,
                                const float* __restrict__ bias,
                                int n_nodes, int Hdim, int do_elu) {
    int i = blockIdx.x * blockDim.x + threadIdx.x;
    if (i >= n_nodes * Hdim) return;
    float v = agg[i] + bias[i % Hdim];
    if (do_elu) v = (v > 0.0f) ? v : (__expf(v) - 1.0f);
    agg[i] = v;
}

// fused layer-2 input written directly as bf16 (GEMM A operand)
__global__ void fuse_bf16_kernel(const float* __restrict__ a,
                                 const float* __restrict__ b,
                                 unsigned short* __restrict__ out, int n) {
    int i = blockIdx.x * blockDim.x + threadIdx.x;
    if (i < n) out[i] = f2bf(0.5f * (a[i] + b[i]));   // ALPHA = 0.5
}

// ---------------------------------------------------------------------------

static void run_gat_conv(const float* Hfeat, int Hdim,
                         const long long* e_src, const long long* e_dst,
                         const float* a_src, const float* a_dst,
                         float* ssrc, float* sdst, unsigned* mord, float* den,
                         float* agg, hipStream_t stream) {
    const int ET = NEDGES + NNODES;
    scores_kernel<<<(NNODES + 7) / 8, 256, 0, stream>>>(
        Hfeat, a_src, a_dst, ssrc, sdst, NNODES, Hdim);
    init_max_kernel<<<(NNODES + 255) / 256, 256, 0, stream>>>(mord, NNODES);
    hipMemsetAsync(den, 0, (size_t)NNODES * sizeof(float), stream);
    hipMemsetAsync(agg, 0, (size_t)NNODES * Hdim * sizeof(float), stream);
    edge_max_kernel<<<(ET + 255) / 256, 256, 0, stream>>>(
        e_src, e_dst, NEDGES, NNODES, ssrc, sdst, mord);
    edge_sum_kernel<<<(ET + 255) / 256, 256, 0, stream>>>(
        e_src, e_dst, NEDGES, NNODES, ssrc, sdst, mord, den);
    edge_scatter_kernel<<<(ET + 7) / 8, 256, 0, stream>>>(
        e_src, e_dst, NEDGES, NNODES, ssrc, sdst, mord, den, Hfeat, agg, Hdim);
}

extern "C" void kernel_launch(void* const* d_in, const int* in_sizes, int n_in,
                              void* d_out, int out_size, void* d_ws, size_t ws_size,
                              hipStream_t stream) {
    const float*     x      = (const float*)d_in[0];
    const long long* sp_idx = (const long long*)d_in[1];
    const long long* gs_idx = (const long long*)d_in[2];
    const float* W_sp = (const float*)d_in[3];
    const float* a_src_sp = (const float*)d_in[4];
    const float* a_dst_sp = (const float*)d_in[5];
    const float* b_sp = (const float*)d_in[6];
    const float* W_gs = (const float*)d_in[7];
    const float* a_src_gs = (const float*)d_in[8];
    const float* a_dst_gs = (const float*)d_in[9];
    const float* b_gs = (const float*)d_in[10];
    const float* W_fu = (const float*)d_in[11];
    const float* a_src_fu = (const float*)d_in[12];
    const float* a_dst_fu = (const float*)d_in[13];
    const float* b_fu = (const float*)d_in[14];
    float* out = (float*)d_out;

    // ---- workspace carve -------------------------------------------------
    char* w = (char*)d_ws;
    const size_t big = (size_t)NNODES * H1 * sizeof(float);     // 102.4 MB
    unsigned short* Xb = (unsigned short*)w;                    // bf16 x / fused
    w += (size_t)NNODES * K1 * sizeof(unsigned short);          // 102.4 MB
    float* B0 = (float*)w; w += big;     // GEMM f32 output h
    float* B1 = (float*)w; w += big;     // agg spatial -> later h_fu (N x 64)
    float* B2 = (float*)w; w += big;     // agg gene
    float* ssrc = (float*)w; w += (size_t)NNODES * sizeof(float);
    float* sdst = (float*)w; w += (size_t)NNODES * sizeof(float);
    float* den  = (float*)w; w += (size_t)NNODES * sizeof(float);
    unsigned* mord = (unsigned*)w; w += (size_t)NNODES * sizeof(unsigned);
    unsigned short* Wp = (unsigned short*)w;   // max 16*16*32*16 bf16 = 256 KB

    const long long* sp_src = sp_idx;
    const long long* sp_dst = sp_idx + NEDGES;
    const long long* gs_src = gs_idx;
    const long long* gs_dst = gs_idx + NEDGES;

    // ---- x -> bf16 once --------------------------------------------------
    {
        int n4 = NNODES * K1 / 4;
        cvt_bf16_kernel<<<(n4 + 255) / 256, 256, 0, stream>>>(x, Xb, n4);
    }

    // ---- layer 1, spatial GAT (512 -> 256), agg into B1 ------------------
    {
        int npack = (K1 / 32) * (H1 / 16) * 32;
        pack_w_kernel<<<(npack + 255) / 256, 256, 0, stream>>>(W_sp, Wp, K1, H1);
        gemm_bf16_kernel<K1, H1><<<dim3(NNODES / 16, (H1 / 16) / 4), 128, 0, stream>>>(
            Xb, Wp, B0);
        run_gat_conv(B0, H1, sp_src, sp_dst, a_src_sp, a_dst_sp,
                     ssrc, sdst, mord, den, B1, stream);
        bias_elu_kernel<<<(NNODES * H1 + 255) / 256, 256, 0, stream>>>(
            B1, b_sp, NNODES, H1, 1);
    }

    // ---- layer 1, gene-sim GAT (512 -> 256), agg into B2 -----------------
    {
        int npack = (K1 / 32) * (H1 / 16) * 32;
        pack_w_kernel<<<(npack + 255) / 256, 256, 0, stream>>>(W_gs, Wp, K1, H1);
        gemm_bf16_kernel<K1, H1><<<dim3(NNODES / 16, (H1 / 16) / 4), 128, 0, stream>>>(
            Xb, Wp, B0);
        run_gat_conv(B0, H1, gs_src, gs_dst, a_src_gs, a_dst_gs,
                     ssrc, sdst, mord, den, B2, stream);
        bias_elu_kernel<<<(NNODES * H1 + 255) / 256, 256, 0, stream>>>(
            B2, b_gs, NNODES, H1, 1);
    }

    // ---- fuse: Xb(bf16, N x 256) = 0.5*B1 + 0.5*B2 -----------------------
    fuse_bf16_kernel<<<(NNODES * H1 + 255) / 256, 256, 0, stream>>>(
        B1, B2, Xb, NNODES * H1);

    // ---- layer 2, fusion GAT (256 -> 64) over spatial edges --------------
    {
        int npack = (H1 / 32) * (H2 / 16) * 32;
        pack_w_kernel<<<(npack + 255) / 256, 256, 0, stream>>>(W_fu, Wp, H1, H2);
        gemm_bf16_kernel<H1, H2><<<dim3(NNODES / 16, (H2 / 16) / 4), 128, 0, stream>>>(
            Xb, Wp, B1);                                // h_fu in B1 (N x 64)
        run_gat_conv(B1, H2, sp_src, sp_dst, a_src_fu, a_dst_fu,
                     ssrc, sdst, mord, den, out, stream);
        bias_elu_kernel<<<(NNODES * H2 + 255) / 256, 256, 0, stream>>>(
            out, b_fu, NNODES, H2, 0);
    }
    (void)in_sizes; (void)n_in; (void)out_size; (void)ws_size;
}
